// ViewGeoAwareTransformer_1185410973965
// MI455X (gfx1250) — matrix operand order repaired and checked by reading, not compiled, not run on hardware
//
#include <hip/hip_runtime.h>
#include <hip/hip_bf16.h>

// ---------------------------------------------------------------------------
// Problem constants (from reference setup_inputs)
// ---------------------------------------------------------------------------
#define BB   2
#define NN   2048
#define MM   2048
#define CIN  128
#define DD   256
#define PH   64
#define DH   1024
#define KNN  16
#define EPSB 1e-5f

typedef __attribute__((ext_vector_type(16))) __bf16 bf16x16;
typedef __attribute__((ext_vector_type(4)))  __bf16 bf16x4;
typedef __attribute__((ext_vector_type(8)))  float  f32x8;

// ---------------------------------------------------------------------------
// WMMA fragment helpers (wave32, v_wmma_f32_16x16x32_bf16)
//   A (16x32): lane L -> row L%16; elem j -> K=(L/16)*16+j
//   B (32x16): lane L -> col L%16; elem j -> K=(L/16)*16+j
//   C (16x16): lane L -> col L%16, rows (L/16)*8 + r
// Swizzled storage: fragment = 32 lanes x 16 contiguous elems per lane (512).
// ---------------------------------------------------------------------------
__device__ __forceinline__ f32x8 wmma_bf16(bf16x16 a, bf16x16 b, f32x8 c) {
  return __builtin_amdgcn_wmma_f32_16x16x32_bf16(false, a, false, b, (short)0, c,
                                                 false, false);
}

__device__ __forceinline__ bf16x16 ld_frag(const __bf16* base, size_t frag) {
  return *(const bf16x16*)(base + (frag * 32 + (threadIdx.x & 31)) * 16);
}

// B-fragment swizzled element address for (kdim c, col k), k-tiles of 32
__device__ __forceinline__ int swz_b(int c, int k) {
  return (((c >> 5) << 5) + (c & 16) + k) * 16 + (c & 15);
}
// C-fragment swizzled element address for (row c, col k), row-tiles of 16
__device__ __forceinline__ int swz_c(int c, int k) {
  return (((c >> 4) << 5) + (((c >> 3) & 1) << 4) + k) * 8 + (c & 7);
}
// A-fragment in-tile (lane, elem) for kdim kt in [0,32): lane=row+16*h, elem=j
__device__ __forceinline__ int swz_a(int row, int kt) {
  int h = (kt >> 4) & 1;
  int j = kt & 15;
  return (row + (h << 4)) * 16 + j;
}

// B-fragment loader from f32 (used only for the tiny sPH tile in geo)
__device__ __forceinline__ bf16x16 ld_b_f32(const float* src, int ldb) {
  const int lane = threadIdx.x & 31;
  const int col  = lane & 15;
  const int kb   = (lane >> 4) << 4;
  bf16x16 b;
#pragma unroll
  for (int j = 0; j < 16; ++j) b[j] = (__bf16)src[(size_t)(kb + j) * ldb + col];
  return b;
}

__device__ __forceinline__ int c_row(int r) {
  return (((threadIdx.x & 31) >> 4) << 3) + r;
}
__device__ __forceinline__ int c_col() { return threadIdx.x & 15; }

// ---------------------------------------------------------------------------
// One-time swizzles: all produce fragment-native bf16
// ---------------------------------------------------------------------------
// Row-major f32 (R x C) -> A-fragment layout; frag = tr*(C/32)+tk
__global__ __launch_bounds__(32)
void swizzleA_kernel(const float* __restrict__ W, __bf16* __restrict__ out, int C) {
  int tk = blockIdx.x, tr = blockIdx.y;
  int lane = threadIdx.x & 31;
  int row = lane & 15;
  int k0 = (lane >> 4) << 4;
  const float* p = W + (size_t)(tr * 16 + row) * C + tk * 32;
  __bf16* o = out + (((size_t)tr * gridDim.x + tk) * 32 + lane) * 16;
#pragma unroll
  for (int j = 0; j < 4; ++j) {
    o[2 * j]         = (__bf16)p[k0 + 2 * j];
    o[2 * j + 1]     = (__bf16)p[k0 + 2 * j + 1];
    o[8 + 2 * j]     = (__bf16)p[8 + k0 + 2 * j];
    o[8 + 2 * j + 1] = (__bf16)p[8 + k0 + 2 * j + 1];
  }
}

// f32 X (B, C, N) -> B-fragment layout; frag = (b*(N/16)+nt)*(C/32)+ck
__global__ __launch_bounds__(32)
void swizzleB_kernel(const float* __restrict__ X, __bf16* __restrict__ out,
                     int C, int N) {
  int nt = blockIdx.x, ck = blockIdx.y, b = blockIdx.z;
  int lane = threadIdx.x & 31;
  int col = lane & 15;
  int kb = (lane >> 4) << 4;
  const float* Xb = X + (size_t)b * C * N;
  __bf16* o = out + ((((size_t)b * gridDim.x + nt) * gridDim.y + ck) * 32 + lane) * 16;
#pragma unroll
  for (int j = 0; j < 16; ++j)
    o[j] = (__bf16)Xb[(size_t)(ck * 32 + kb + j) * N + nt * 16 + col];
}

// f32 Q (B, C, M) -> A-fragment layout of Q^T (M rows, C kdims);
// frag = (b*(M/16)+rt)*(C/32)+tk
__global__ __launch_bounds__(32)
void swizzleAT_kernel(const float* __restrict__ Q, __bf16* __restrict__ out,
                      int C, int M) {
  int tk = blockIdx.x, rt = blockIdx.y, b = blockIdx.z;
  int lane = threadIdx.x & 31;
  int row = lane & 15;
  int k0 = (lane >> 4) << 4;
  const float* Qb = Q + (size_t)b * C * M + (size_t)rt * 16 + row;
  __bf16* o = out + ((((size_t)b * gridDim.y + rt) * gridDim.x + tk) * 32 + lane) * 16;
#pragma unroll
  for (int j = 0; j < 4; ++j) {
    int ka = tk * 32 + k0 + 2 * j;
    int kc = tk * 32 + 8 + k0 + 2 * j;
    o[2 * j]         = (__bf16)Qb[(size_t)ka * M];
    o[2 * j + 1]     = (__bf16)Qb[(size_t)(ka + 1) * M];
    o[8 + 2 * j]     = (__bf16)Qb[(size_t)kc * M];
    o[8 + 2 * j + 1] = (__bf16)Qb[(size_t)(kc + 1) * M];
  }
}

// Fold BN (+ conv bias) into per-channel scale/shift
__global__ void bnfold_kernel(const float* g, const float* be, const float* m,
                              const float* v, const float* b1,
                              float* inv, float* sh, int C) {
  int i = blockIdx.x * blockDim.x + threadIdx.x;
  if (i >= C) return;
  float iv = g[i] * rsqrtf(v[i] + EPSB);
  inv[i] = iv;
  sh[i] = b1[i] * iv + be[i] - m[i] * iv;
}

// ---------------------------------------------------------------------------
// Column L2 norms of X (B,C,N) -> out (B,N)
// ---------------------------------------------------------------------------
__global__ void colnorm_kernel(const float* __restrict__ X, float* __restrict__ out,
                               int C, int N) {
  int b = blockIdx.y;
  int n = blockIdx.x * blockDim.x + threadIdx.x;
  if (n >= N) return;
  const float* p = X + (size_t)b * C * N + n;
  float s = 0.f;
  for (int c = 0; c < C; ++c) { float x = p[(size_t)c * N]; s += x * x; }
  out[b * N + n] = sqrtf(s);
}

// ---------------------------------------------------------------------------
// Yt(b,N,Dout) = (W * X + bias)^T : all-fragment GEMM, transposed store so
// later gathers read contiguous rows.
// ---------------------------------------------------------------------------
__global__ __launch_bounds__(32)
void proj_kernel(const __bf16* __restrict__ Ws, const float* __restrict__ bias,
                 const __bf16* __restrict__ Xs, float* __restrict__ Yt,
                 int Dout, int Cin, int N) {
  int b = blockIdx.z, ot = blockIdx.y, nt = blockIdx.x;
  const int kt = Cin >> 5;
  f32x8 acc = {};
  for (int c0 = 0; c0 < kt; ++c0)
    acc = wmma_bf16(ld_frag(Ws, ot * kt + c0),
                    ld_frag(Xs, ((size_t)b * (N >> 4) + nt) * kt + c0), acc);
  int col = c_col();
#pragma unroll
  for (int r = 0; r < 8; ++r) {
    int row = ot * 16 + c_row(r);
    Yt[((size_t)b * N + nt * 16 + col) * Dout + row] = acc[r] + bias[row];
  }
}

// ---------------------------------------------------------------------------
// Normalized cosine matrix (all-fragment GEMM)
// ---------------------------------------------------------------------------
__global__ __launch_bounds__(32)
void cos_kernel(const __bf16* __restrict__ qTs, const __bf16* __restrict__ kpcs,
                const float* __restrict__ nq, const float* __restrict__ nk,
                float* __restrict__ cosm) {
  int b = blockIdx.z, mt = blockIdx.y, nt = blockIdx.x;
  f32x8 acc = {};
  for (int tk = 0; tk < CIN / 32; ++tk)
    acc = wmma_bf16(ld_frag(qTs, ((size_t)b * (MM >> 4) + mt) * (CIN / 32) + tk),
                    ld_frag(kpcs, ((size_t)b * (NN >> 4) + nt) * (CIN / 32) + tk),
                    acc);
  int col = c_col();
  float invk = 1.f / nk[b * NN + nt * 16 + col];
  float* Cb = cosm + (size_t)b * MM * NN + (size_t)mt * 16 * NN + nt * 16;
#pragma unroll
  for (int r = 0; r < 8; ++r) {
    int row = c_row(r);
    Cb[(size_t)row * NN + col] = acc[r] * invk / nq[b * MM + mt * 16 + row];
  }
}

// ---------------------------------------------------------------------------
// Top-16 (iterative argmax, smallest index wins ties)
// ---------------------------------------------------------------------------
#define TKT 256
__device__ __forceinline__ void select_topk(float* srow, float* sv, int* si,
                                            int* out_idx, int N) {
  for (int kk = 0; kk < KNN; ++kk) {
    float bv = -__builtin_inff(); int bi = N;
    for (int i = threadIdx.x; i < N; i += TKT) {
      float x = srow[i];
      if (x > bv || (x == bv && i < bi)) { bv = x; bi = i; }
    }
    sv[threadIdx.x] = bv; si[threadIdx.x] = bi;
    __syncthreads();
    for (int s = TKT / 2; s > 0; s >>= 1) {
      if ((int)threadIdx.x < s) {
        float ov = sv[threadIdx.x + s]; int oi = si[threadIdx.x + s];
        if (ov > sv[threadIdx.x] || (ov == sv[threadIdx.x] && oi < si[threadIdx.x])) {
          sv[threadIdx.x] = ov; si[threadIdx.x] = oi;
        }
      }
      __syncthreads();
    }
    if (threadIdx.x == 0) {
      out_idx[kk] = si[0];
      srow[si[0]] = -__builtin_inff();
    }
    __syncthreads();
  }
}

__global__ void topk_rows_kernel(const float* __restrict__ score,
                                 int* __restrict__ idx, int N) {
  __shared__ float srow[NN];
  __shared__ float sv[TKT];
  __shared__ int   si[TKT];
  int b = blockIdx.y, m = blockIdx.x;
  const float* row = score + ((size_t)b * gridDim.x + m) * N;
  for (int i = threadIdx.x; i < N; i += TKT) srow[i] = row[i];
  __syncthreads();
  select_topk(srow, sv, si, idx + ((size_t)b * gridDim.x + m) * KNN, N);
}

__global__ void topk_geo_kernel(const float* __restrict__ pos,
                                int* __restrict__ idx, int N) {
  __shared__ float spos[3 * NN];
  __shared__ float srow[NN];
  __shared__ float sv[TKT];
  __shared__ int   si[TKT];
  int b = blockIdx.y, n = blockIdx.x;
  const float* pb = pos + (size_t)b * 3 * N;
  for (int i = threadIdx.x; i < 3 * N; i += TKT) spos[i] = pb[i];
  __syncthreads();
  float px = spos[n], py = spos[N + n], pz = spos[2 * N + n];
  for (int i = threadIdx.x; i < N; i += TKT) {
    float dx = px - spos[i], dy = py - spos[N + i], dz = pz - spos[2 * N + i];
    srow[i] = -(dx * dx + dy * dy + dz * dz);
  }
  __syncthreads();
  select_topk(srow, sv, si, idx + ((size_t)b * gridDim.x + n) * KNN, N);
}

// ---------------------------------------------------------------------------
// Softmax stats over m of cos^T per (b,n)
// ---------------------------------------------------------------------------
__global__ void rowstats_kernel(const float* __restrict__ cosm,
                                float* __restrict__ rmax, float* __restrict__ rsum,
                                int M, int N) {
  int b = blockIdx.y;
  int n = blockIdx.x * blockDim.x + threadIdx.x;
  if (n >= N) return;
  const float* p = cosm + (size_t)b * M * N + n;
  float mx = -__builtin_inff();
  for (int m = 0; m < M; ++m) mx = fmaxf(mx, p[(size_t)m * N]);
  float s = 0.f;
  for (int m = 0; m < M; ++m) s += __expf(p[(size_t)m * N] - mx);
  rmax[b * N + n] = mx;
  rsum[b * N + n] = s;
}

// exp(cos - rmax) -> bf16 in B-fragment layout over (Kdim=M, cols=N)
// frag = (b*(N/16)+nt)*(M/32)+mt
__global__ void expcos_kernel(const float* __restrict__ cosm,
                              const float* __restrict__ rmax,
                              __bf16* __restrict__ expB) {
  int b = blockIdx.z, m = blockIdx.y;
  int n = blockIdx.x * blockDim.x + threadIdx.x;
  float e = __expf(cosm[((size_t)b * MM + m) * NN + n] - rmax[b * NN + n]);
  int kt = m & 31;
  size_t frag = ((size_t)b * (NN >> 4) + (n >> 4)) * (MM >> 5) + (m >> 5);
  expB[(frag * 32 + (kt & 16) + (n & 15)) * 16 + (kt & 15)] = (__bf16)e;
}

// ---------------------------------------------------------------------------
// Big fused MLP over one 256x16 tile: accumulators live in 128 VGPRs,
// A: 2x global b128/lane, B: 2x ds b128/lane, one shared B frag per 16 WMMAs.
// ---------------------------------------------------------------------------
__device__ void mlp_big(const __bf16* sIn, float* sAcc, __bf16* sH,
                        const __bf16* W1s, const float* inv1, const float* sh1,
                        const __bf16* W2s) {
  const int lane = threadIdx.x & 31;
  const int col  = lane & 15;
  const int rb   = (lane >> 4) << 3;
  f32x8 acc[16];
#pragma unroll
  for (int ot = 0; ot < 16; ++ot) acc[ot] = (f32x8){};
  for (int hc = 0; hc < DH; hc += 32) {
#pragma unroll
    for (int sub = 0; sub < 2; ++sub) {
      const int rt = (hc >> 4) + sub;
      f32x8 h = {};
#pragma unroll
      for (int c0 = 0; c0 < 8; ++c0)
        h = wmma_bf16(ld_frag(W1s, rt * 8 + c0), ld_frag(sIn, c0), h);
#pragma unroll
      for (int r = 0; r < 8; ++r) {
        int ch = hc + 16 * sub + rb + r;
        float x = h[r] * inv1[ch] + sh1[ch];
        sH[(16 * sub + col) * 16 + rb + r] = (__bf16)fmaxf(x, 0.f);
      }
    }
    __syncthreads();
    bf16x16 hb = ld_frag(sH, 0);
#pragma unroll
    for (int ot = 0; ot < 16; ++ot)
      acc[ot] = wmma_bf16(ld_frag(W2s, ot * 32 + (hc >> 5)), hb, acc[ot]);
    __syncthreads();
  }
#pragma unroll
  for (int ot = 0; ot < 16; ++ot)
    *(f32x8*)(sAcc + ((size_t)ot * 32 + lane) * 8) = acc[ot];
  __syncthreads();
}

// ---------------------------------------------------------------------------
// View branch: float4 gathers from keyT/valT -> MLP -> softmax over K ->
// weighted sum; agg written directly in A-fragment bf16 layout.
// ---------------------------------------------------------------------------
__global__ __launch_bounds__(32)
void view_attn_kernel(const float* __restrict__ keyT, const float* __restrict__ valT,
                      const int* __restrict__ idxv,
                      const __bf16* __restrict__ vW1s,
                      const float* __restrict__ vinv, const float* __restrict__ vsh,
                      const __bf16* __restrict__ vW2s, const float* __restrict__ vb2,
                      __bf16* __restrict__ aggA) {
  __shared__ __align__(32) __bf16 sIn[DD * 16];
  __shared__ __align__(32) float  sVal[KNN * DD];   // [k][c]
  __shared__ __align__(32) float  sAcc[DD * 16];
  __shared__ __align__(32) __bf16 sH[32 * 16];
  __shared__ int sIdx[KNN];
  int b = blockIdx.y, m = blockIdx.x, lane = threadIdx.x;
  if (lane < KNN) sIdx[lane] = idxv[((size_t)b * gridDim.x + m) * KNN + lane];
  __syncthreads();
  const float4* keyT4 = (const float4*)(keyT + (size_t)b * NN * DD);
  const float4* valT4 = (const float4*)(valT + (size_t)b * NN * DD);
  float4* sVal4 = (float4*)sVal;
  for (int e = lane; e < KNN * (DD / 4); e += 32) {
    int k = e >> 6, c4 = e & 63;
    int j = sIdx[k];
    float4 kv = keyT4[(size_t)j * (DD / 4) + c4];
    float4 vv = valT4[(size_t)j * (DD / 4) + c4];
    int c = c4 << 2;
    bf16x4 pk;
    pk[0] = (__bf16)kv.x; pk[1] = (__bf16)kv.y;
    pk[2] = (__bf16)kv.z; pk[3] = (__bf16)kv.w;
    *(bf16x4*)&sIn[swz_b(c, k)] = pk;
    sVal4[k * (DD / 4) + c4] = vv;
  }
  __syncthreads();
  mlp_big(sIn, sAcc, sH, vW1s, vinv, vsh, vW2s);
  // softmax over K + weighted sum; write agg in A-fragment layout (kdim = m)
  int kt = m & 31;
  size_t fbase = ((size_t)b * (DD >> 4)) * (MM >> 5) + (m >> 5);
  int apos = swz_a(0, kt);
  for (int c = lane; c < DD; c += 32) {
    float bias = vb2[c];
    float mx = -__builtin_inff();
#pragma unroll
    for (int k = 0; k < KNN; ++k) mx = fmaxf(mx, sAcc[swz_c(c, k)] + bias);
    float s = 0.f, out = 0.f;
#pragma unroll
    for (int k = 0; k < KNN; ++k) {
      float e = __expf(sAcc[swz_c(c, k)] + bias - mx);
      s += e;
      out += e * sVal[k * DD + c];
    }
    size_t frag = fbase + (size_t)(c >> 4) * (MM >> 5);
    aggA[frag * 512 + apos + (c & 15) * 16] = (__bf16)(out / s);
  }
}

// ---------------------------------------------------------------------------
// feat_view = softmax_m(cos^T) * agg ; all-fragment GEMM, result stored as
// B-fragment bf16 for the final projection.
// ---------------------------------------------------------------------------
__global__ __launch_bounds__(32)
void featview_kernel(const __bf16* __restrict__ aggA, const __bf16* __restrict__ expB,
                     const float* __restrict__ rsum, __bf16* __restrict__ fviewB) {
  int b = blockIdx.z, dt = blockIdx.y, nt = blockIdx.x;
  f32x8 acc = {};
  for (int mt = 0; mt < MM / 32; ++mt)
    acc = wmma_bf16(ld_frag(aggA, ((size_t)b * (DD >> 4) + dt) * (MM >> 5) + mt),
                    ld_frag(expB, ((size_t)b * (NN >> 4) + nt) * (MM >> 5) + mt),
                    acc);
  int col = c_col();
  float inv = 1.f / rsum[b * NN + nt * 16 + col];
#pragma unroll
  for (int r = 0; r < 8; ++r) {
    int d = dt * 16 + c_row(r);
    size_t frag = ((size_t)b * (NN >> 4) + nt) * (DD >> 5) + (d >> 5);
    fviewB[(frag * 32 + (d & 16) + col) * 16 + (d & 15)] = (__bf16)(acc[r] * inv);
  }
}

// ---------------------------------------------------------------------------
// Geo branch: pe first (WMMA layer2), then fused gather of key_geo+pe,
// big MLP, softmax, weighted sum; feat stored as B-fragment bf16.
// ---------------------------------------------------------------------------
__global__ __launch_bounds__(32)
void geo_attn_kernel(const float* __restrict__ keyT, const float* __restrict__ valT,
                     const float* __restrict__ pos, const int* __restrict__ idxg,
                     const float* pW1, const float* pinv, const float* psh,
                     const __bf16* __restrict__ pW2s, const float* pb2,
                     const __bf16* __restrict__ gW1s,
                     const float* __restrict__ ginv, const float* __restrict__ gsh,
                     const __bf16* __restrict__ gW2s, const float* __restrict__ gb2,
                     __bf16* __restrict__ fgeoB) {
  __shared__ __align__(32) __bf16 sIn[DD * 16];
  __shared__ __align__(32) float  sPE[KNN * DD];    // [k][c]
  __shared__ __align__(32) float  sAcc[DD * 16];
  __shared__ __align__(32) float  sPH[PH * 16];     // [hh][k]
  __shared__ __align__(32) float  sKn[DD];
  __shared__ __align__(32) __bf16 sH[32 * 16];
  __shared__ float sPR[3 * 16];
  __shared__ int sIdx[KNN];
  int b = blockIdx.y, n = blockIdx.x, lane = threadIdx.x;
  if (lane < KNN) sIdx[lane] = idxg[((size_t)b * gridDim.x + n) * KNN + lane];
  __syncthreads();
  const float4* keyT4 = (const float4*)(keyT + (size_t)b * NN * DD);
  const float* posb = pos + (size_t)b * 3 * NN;
  float4* sKn4 = (float4*)sKn;
  for (int i = lane; i < DD / 4; i += 32)
    sKn4[i] = keyT4[(size_t)n * (DD / 4) + i];
  for (int e = lane; e < 3 * 16; e += 32) {
    int c = e >> 4, k = e & 15;
    sPR[e] = posb[(size_t)c * NN + n] - posb[(size_t)c * NN + sIdx[k]];
  }
  __syncthreads();
  // pe layer 1 (K=3, VALU), BN+ReLU folded
  for (int e = lane; e < PH * 16; e += 32) {
    int hh = e >> 4, k = e & 15;
    float h = pW1[hh * 3 + 0] * sPR[k] + pW1[hh * 3 + 1] * sPR[16 + k] +
              pW1[hh * 3 + 2] * sPR[32 + k];
    sPH[e] = fmaxf(h * pinv[hh] + psh[hh], 0.f);
  }
  __syncthreads();
  // pe layer 2 via WMMA; write sPE in [k][c] layout
  {
    const int col = lane & 15;
    const int rb  = (lane >> 4) << 3;
    for (int ot = 0; ot < 16; ++ot) {
      f32x8 acc = {};
#pragma unroll
      for (int kc = 0; kc < 2; ++kc)
        acc = wmma_bf16(ld_frag(pW2s, ot * 2 + kc),
                        ld_b_f32(sPH + kc * 32 * 16, 16), acc);
#pragma unroll
      for (int r = 0; r < 8; ++r) {
        int c = ot * 16 + rb + r;
        sPE[col * DD + c] = acc[r] + pb2[c];
      }
    }
  }
  __syncthreads();
  // t = key_geo + pe  (float4 gathers, packed bf16x4 LDS stores)
  for (int e = lane; e < KNN * (DD / 4); e += 32) {
    int k = e >> 6, c4 = e & 63;
    int j = sIdx[k];
    float4 kj = keyT4[(size_t)j * (DD / 4) + c4];
    float4 kn = sKn4[c4];
    int c = c4 << 2;
    const float* pe4 = &sPE[k * DD + c];
    bf16x4 pk;
    pk[0] = (__bf16)(kn.x - kj.x + pe4[0]);
    pk[1] = (__bf16)(kn.y - kj.y + pe4[1]);
    pk[2] = (__bf16)(kn.z - kj.z + pe4[2]);
    pk[3] = (__bf16)(kn.w - kj.w + pe4[3]);
    *(bf16x4*)&sIn[swz_b(c, k)] = pk;
  }
  __syncthreads();
  mlp_big(sIn, sAcc, sH, gW1s, ginv, gsh, gW2s);
  const float* valn = valT + ((size_t)b * NN + n) * DD;
  for (int c = lane; c < DD; c += 32) {
    float bias = gb2[c];
    float vn = valn[c];
    float mx = -__builtin_inff();
#pragma unroll
    for (int k = 0; k < KNN; ++k) mx = fmaxf(mx, sAcc[swz_c(c, k)] + bias);
    float s = 0.f, out = 0.f;
#pragma unroll
    for (int k = 0; k < KNN; ++k) {
      float e = __expf(sAcc[swz_c(c, k)] + bias - mx);
      s += e;
      out += e * (vn + sPE[k * DD + c]);
    }
    size_t frag = ((size_t)b * (NN >> 4) + (n >> 4)) * (DD >> 5) + (c >> 5);
    fgeoB[(frag * 32 + (c & 16) + (n & 15)) * 16 + (c & 15)] = (__bf16)(out / s);
  }
}

// ---------------------------------------------------------------------------
// out = value + Wvo*feat_view + bvo + Wgo*feat_geo + bgo  (all-fragment GEMM)
// ---------------------------------------------------------------------------
__global__ __launch_bounds__(32)
void final_kernel(const float* __restrict__ value,
                  const __bf16* __restrict__ Wvos, const float* __restrict__ bvo,
                  const __bf16* __restrict__ Wgos, const float* __restrict__ bgo,
                  const __bf16* __restrict__ fviewB, const __bf16* __restrict__ fgeoB,
                  float* __restrict__ out) {
  int b = blockIdx.z, ct = blockIdx.y, nt = blockIdx.x;
  const size_t bfrag = ((size_t)b * (NN >> 4) + nt) * (DD >> 5);
  f32x8 acc = {};
#pragma unroll
  for (int c0 = 0; c0 < DD / 32; ++c0)
    acc = wmma_bf16(ld_frag(Wvos, ct * (DD / 32) + c0), ld_frag(fviewB, bfrag + c0),
                    acc);
#pragma unroll
  for (int c0 = 0; c0 < DD / 32; ++c0)
    acc = wmma_bf16(ld_frag(Wgos, ct * (DD / 32) + c0), ld_frag(fgeoB, bfrag + c0),
                    acc);
  int col = c_col();
#pragma unroll
  for (int r = 0; r < 8; ++r) {
    int row = ct * 16 + c_row(r);
    size_t o = ((size_t)b * CIN + row) * NN + nt * 16 + col;
    out[o] = acc[r] + value[o] + bvo[row] + bgo[row];
  }
}

// ---------------------------------------------------------------------------
// Host launcher
// ---------------------------------------------------------------------------
extern "C" void kernel_launch(void* const* d_in, const int* in_sizes, int n_in,
                              void* d_out, int out_size, void* d_ws, size_t ws_size,
                              hipStream_t stream) {
  const float* pos      = (const float*)d_in[0];
  const float* query_im = (const float*)d_in[1];
  const float* key_pc   = (const float*)d_in[2];
  const float* value    = (const float*)d_in[3];
  const float* Wk  = (const float*)d_in[4];  const float* bk  = (const float*)d_in[5];
  const float* Wv  = (const float*)d_in[6];  const float* bv  = (const float*)d_in[7];
  const float* pW1 = (const float*)d_in[8];  const float* pb1 = (const float*)d_in[9];
  const float* pW2 = (const float*)d_in[10]; const float* pb2 = (const float*)d_in[11];
  const float* vW1 = (const float*)d_in[12]; const float* vb1 = (const float*)d_in[13];
  const float* vW2 = (const float*)d_in[14]; const float* vb2 = (const float*)d_in[15];
  const float* gW1 = (const float*)d_in[16]; const float* gb1 = (const float*)d_in[17];
  const float* gW2 = (const float*)d_in[18]; const float* gb2 = (const float*)d_in[19];
  const float* Wvo = (const float*)d_in[20]; const float* bvo = (const float*)d_in[21];
  const float* Wgo = (const float*)d_in[22]; const float* bgo = (const float*)d_in[23];
  const float* pg  = (const float*)d_in[24]; const float* pbe = (const float*)d_in[25];
  const float* pm  = (const float*)d_in[26]; const float* pvv = (const float*)d_in[27];
  const float* vg  = (const float*)d_in[28]; const float* vbe = (const float*)d_in[29];
  const float* vm  = (const float*)d_in[30]; const float* vvv = (const float*)d_in[31];
  const float* gg  = (const float*)d_in[32]; const float* gbe = (const float*)d_in[33];
  const float* gm  = (const float*)d_in[34]; const float* gvv = (const float*)d_in[35];
  float* outp = (float*)d_out;

  // ---- workspace layout: f32, then int, then bf16 (all 32B aligned) ----
  float* w     = (float*)d_ws;
  float* keyT  = w;                                  // B*N*D
  float* valT  = keyT + (size_t)BB * NN * DD;
  float* cosm  = valT + (size_t)BB * NN * DD;        // B*M*N
  float* nq    = cosm + (size_t)BB * MM * NN;
  float* nk    = nq + (size_t)BB * MM;
  float* rmx   = nk + (size_t)BB * NN;
  float* rsm   = rmx + (size_t)BB * NN;
  float* vinv  = rsm + (size_t)BB * NN;
  float* vsh   = vinv + DH;
  float* ginv  = vsh + DH;
  float* gsh   = ginv + DH;
  float* pinv  = gsh + DH;   // PH used, padded to 1024
  float* psh   = pinv + 1024;
  int*   idxv  = (int*)(psh + 1024);
  int*   idxg  = idxv + (size_t)BB * MM * KNN;
  __bf16* Wks  = (__bf16*)(idxg + (size_t)BB * NN * KNN);
  __bf16* Wvs  = Wks + (size_t)DD * CIN;
  __bf16* Wvos = Wvs + (size_t)DD * CIN;
  __bf16* Wgos = Wvos + (size_t)CIN * DD;
  __bf16* pW2s = Wgos + (size_t)CIN * DD;            // 256*64
  __bf16* vW1s = pW2s + (size_t)DD * PH;
  __bf16* vW2s = vW1s + (size_t)DH * DD;
  __bf16* gW1s = vW2s + (size_t)DD * DH;
  __bf16* gW2s = gW1s + (size_t)DH * DD;
  __bf16* kpcs = gW2s + (size_t)DD * DH;             // B*CIN*NN (B-frag)
  __bf16* vals = kpcs + (size_t)BB * CIN * NN;
  __bf16* qTs  = vals + (size_t)BB * CIN * NN;       // B*MM*CIN (A-frag of Q^T)
  __bf16* aggA = qTs + (size_t)BB * MM * CIN;        // B*DD*MM (A-frag)
  __bf16* expB = aggA + (size_t)BB * DD * MM;        // B*MM*NN (B-frag)
  __bf16* fvB  = expB + (size_t)BB * MM * NN;        // B*DD*NN (B-frag)
  __bf16* fgB  = fvB + (size_t)BB * DD * NN;

  // 0. one-time swizzles + BN folds
  swizzleA_kernel<<<dim3(CIN / 32, DD / 16), 32, 0, stream>>>(Wk, Wks, CIN);
  swizzleA_kernel<<<dim3(CIN / 32, DD / 16), 32, 0, stream>>>(Wv, Wvs, CIN);
  swizzleA_kernel<<<dim3(DD / 32, CIN / 16), 32, 0, stream>>>(Wvo, Wvos, DD);
  swizzleA_kernel<<<dim3(DD / 32, CIN / 16), 32, 0, stream>>>(Wgo, Wgos, DD);
  swizzleA_kernel<<<dim3(PH / 32, DD / 16), 32, 0, stream>>>(pW2, pW2s, PH);
  swizzleA_kernel<<<dim3(DD / 32, DH / 16), 32, 0, stream>>>(vW1, vW1s, DD);
  swizzleA_kernel<<<dim3(DH / 32, DD / 16), 32, 0, stream>>>(vW2, vW2s, DH);
  swizzleA_kernel<<<dim3(DD / 32, DH / 16), 32, 0, stream>>>(gW1, gW1s, DD);
  swizzleA_kernel<<<dim3(DH / 32, DD / 16), 32, 0, stream>>>(gW2, gW2s, DH);
  swizzleB_kernel<<<dim3(NN / 16, CIN / 32, BB), 32, 0, stream>>>(key_pc, kpcs, CIN, NN);
  swizzleB_kernel<<<dim3(NN / 16, CIN / 32, BB), 32, 0, stream>>>(value, vals, CIN, NN);
  swizzleAT_kernel<<<dim3(CIN / 32, MM / 16, BB), 32, 0, stream>>>(query_im, qTs, CIN, MM);
  bnfold_kernel<<<dim3(DH / 256), 256, 0, stream>>>(vg, vbe, vm, vvv, vb1, vinv, vsh, DH);
  bnfold_kernel<<<dim3(DH / 256), 256, 0, stream>>>(gg, gbe, gm, gvv, gb1, ginv, gsh, DH);
  bnfold_kernel<<<dim3(1), 256, 0, stream>>>(pg, pbe, pm, pvv, pb1, pinv, psh, PH);

  // 1. column norms
  colnorm_kernel<<<dim3(MM / 256, BB), 256, 0, stream>>>(query_im, nq, CIN, MM);
  colnorm_kernel<<<dim3(NN / 256, BB), 256, 0, stream>>>(key_pc, nk, CIN, NN);

  // 2. geo KNN
  topk_geo_kernel<<<dim3(NN, BB), TKT, 0, stream>>>(pos, idxg, NN);

  // 3. key/val projections (transposed outputs for fast gathers)
  proj_kernel<<<dim3(NN / 16, DD / 16, BB), 32, 0, stream>>>(Wks, bk, kpcs, keyT,
                                                             DD, CIN, NN);
  proj_kernel<<<dim3(NN / 16, DD / 16, BB), 32, 0, stream>>>(Wvs, bv, vals, valT,
                                                             DD, CIN, NN);

  // 4. normalized cosine matrix
  cos_kernel<<<dim3(NN / 16, MM / 16, BB), 32, 0, stream>>>(qTs, kpcs, nq, nk, cosm);

  // 5. view KNN over cos rows
  topk_rows_kernel<<<dim3(MM, BB), TKT, 0, stream>>>(cosm, idxv, NN);

  // 6. softmax stats + exp(cos - max) in B-fragment bf16
  rowstats_kernel<<<dim3(NN / 256, BB), 256, 0, stream>>>(cosm, rmx, rsm, MM, NN);
  expcos_kernel<<<dim3(NN / 256, MM, BB), 256, 0, stream>>>(cosm, rmx, expB);

  // 7. fused view attention -> aggA
  view_attn_kernel<<<dim3(MM, BB), 32, 0, stream>>>(keyT, valT, idxv, vW1s, vinv,
                                                    vsh, vW2s, vb2, aggA);

  // 8. feat_view = attn_cos * agg (all-fragment GEMM)
  featview_kernel<<<dim3(NN / 16, DD / 16, BB), 32, 0, stream>>>(aggA, expB, rsm, fvB);

  // 9. fused geo attention -> fgB
  geo_attn_kernel<<<dim3(NN, BB), 32, 0, stream>>>(keyT, valT, pos, idxg,
                                                   pW1, pinv, psh, pW2s, pb2,
                                                   gW1s, ginv, gsh, gW2s, gb2, fgB);

  // 10. residual + output projections
  final_kernel<<<dim3(NN / 16, CIN / 16, BB), 32, 0, stream>>>(value, Wvos, bvo,
                                                               Wgos, bgo, fvB, fgB,
                                                               outp);
}
